// RGCNEncoder_61881888801357
// MI455X (gfx1250) — matrix operand order repaired and checked by reading, not compiled, not run on hardware
//
#include <hip/hip_runtime.h>

typedef __bf16 bf16_t;
typedef __attribute__((ext_vector_type(16))) __bf16 v16bf;
typedef __attribute__((ext_vector_type(8)))  __bf16 v8bf;
typedef __attribute__((ext_vector_type(4)))  __bf16 v4bf;
typedef __attribute__((ext_vector_type(8)))  float  v8f;

#define N_NODES 30000
#define R_REL   8
#define NBASES  30
#define IN_DIM  256
#define Z_DIM   128
#define HID_DIM 256
#define E_EDGES (N_NODES * 16)

#define MT 2   // M tiles per wave
#define NT 4   // N tiles per wave

// ---------------------------------------------------------------------------
// W_t[r][o][k] = sum_b comp[r,b]*bases[b][k][o]  (r < R);  W_t[R][o][k] = root
// Transposed (o-major) so WMMA B-operand loads are contiguous per lane.
// ---------------------------------------------------------------------------
__global__ void prep_w_kernel(const float* __restrict__ bases,
                              const float* __restrict__ comp,
                              const float* __restrict__ root,
                              bf16_t* __restrict__ Wt, int K, int O) {
    int idx = blockIdx.x * blockDim.x + threadIdx.x;
    int total = (R_REL + 1) * O * K;
    if (idx >= total) return;
    int k = idx % K;
    int o = (idx / K) % O;
    int r = idx / (K * O);
    float acc;
    if (r < R_REL) {
        acc = 0.f;
        #pragma unroll 6
        for (int b = 0; b < NBASES; ++b)
            acc += comp[r * NBASES + b] * bases[((size_t)b * K + k) * O + o];
    } else {
        acc = root[(size_t)k * O + o];
    }
    Wt[idx] = (bf16_t)acc;
}

__global__ void cvt_bf16_kernel(const float* __restrict__ in,
                                bf16_t* __restrict__ out, int n) {
    int i = blockIdx.x * blockDim.x + threadIdx.x;
    if (i < n) out[i] = (bf16_t)in[i];
}

__global__ void zero_f32_kernel(float* __restrict__ p, int n) {
    int i = blockIdx.x * blockDim.x + threadIdx.x;
    if (i < n) p[i] = 0.f;
}

__global__ void count_edges_kernel(const int* __restrict__ ei,
                                   const int* __restrict__ et,
                                   float* __restrict__ cnt, int E) {
    int e = blockIdx.x * blockDim.x + threadIdx.x;
    if (e >= E) return;
    int dst = ei[E + e];
    int r   = et[e];
    __hip_atomic_fetch_add(&cnt[dst * R_REL + r], 1.0f,
                           __ATOMIC_RELAXED, __HIP_MEMORY_SCOPE_AGENT);
}

__global__ void invert_cnt_kernel(float* __restrict__ cnt, int n) {
    int i = blockIdx.x * blockDim.x + threadIdx.x;
    if (i < n) cnt[i] = 1.0f / fmaxf(cnt[i], 1.0f);
}

// ---------------------------------------------------------------------------
// Register-blocked WMMA GEMM: each wave computes an MTx16 x NTx16 tile group,
// K=256 in 8 steps of v_wmma_f32_16x16x32_bf16 (64 WMMAs / wave).
// A: [M][K] bf16 row-major; Wt: [(R+1)*O][K] bf16 (output-col major).
// Columns c < R*O -> bf16 xall; c >= R*O -> f32 root transform + bias.
// ---------------------------------------------------------------------------
__global__ void __launch_bounds__(256)
gemm_wmma_kernel(const bf16_t* __restrict__ A, const bf16_t* __restrict__ Bm,
                 bf16_t* __restrict__ xall, float* __restrict__ rootout,
                 const float* __restrict__ bias, int Odim, int tilesC) {
    const int K  = 256;               // both layers have K = 256
    const int RO = R_REL * Odim;
    const int tilesM  = N_NODES / 16;            // 1875
    const int groupsM = (tilesM + MT - 1) / MT;  // 938 (last group clamped)
    const int groupsC = tilesC / NT;
    int wid = blockIdx.x * (blockDim.x >> 5) + (threadIdx.x >> 5);
    if (wid >= groupsM * groupsC) return;
    int gm = wid / groupsC;
    int gc = wid % groupsC;
    int lane = threadIdx.x & 31;
    int half = lane >> 4;             // which half-wave
    int l16  = lane & 15;

    // k order used here (contiguous, A and B alike). A: lanes 0-15 hold K 0..15 of row M=l,
    // lanes 16-31 hold K 16..31.  B (32x16): lanes 0-15 hold K 0..15 of
    // col N=l, lanes 16-31 hold K 16..31.
    const bf16_t* Arow[MT];
    const bf16_t* Bcol[NT];
    #pragma unroll
    for (int mi = 0; mi < MT; ++mi) {
        int tm = gm * MT + mi;
        if (tm > tilesM - 1) tm = tilesM - 1;     // clamp: keep EXEC all-ones
        Arow[mi] = A + (size_t)(tm * 16 + l16) * K;
    }
    #pragma unroll
    for (int ni = 0; ni < NT; ++ni)
        Bcol[ni] = Bm + (size_t)((gc * NT + ni) * 16 + l16) * K;

    v8f acc[MT][NT] = {};
    #pragma unroll
    for (int kk = 0; kk < 256; kk += 32) {
        v16bf a[MT], b[NT];
        #pragma unroll
        for (int mi = 0; mi < MT; ++mi) {
            const v8bf a0 = *(const v8bf*)(Arow[mi] + kk + half * 16);
            const v8bf a1 = *(const v8bf*)(Arow[mi] + kk + half * 16 + 8);
            #pragma unroll
            for (int i = 0; i < 8; ++i) { a[mi][i] = a0[i]; a[mi][i + 8] = a1[i]; }
        }
        #pragma unroll
        for (int ni = 0; ni < NT; ++ni)
            b[ni] = *(const v16bf*)(Bcol[ni] + kk + half * 16);
        #pragma unroll
        for (int mi = 0; mi < MT; ++mi)
            #pragma unroll
            for (int ni = 0; ni < NT; ++ni)
                acc[mi][ni] = __builtin_amdgcn_wmma_f32_16x16x32_bf16(
                    false, a[mi], false, b[ni], (short)0, acc[mi][ni], false, false);
    }

    // C/D layout: VGPR v holds row M = v + 8*half, column N = lane&15.
    #pragma unroll
    for (int mi = 0; mi < MT; ++mi) {
        int tm = gm * MT + mi;
        if (tm >= tilesM) continue;               // wave-uniform skip
        #pragma unroll
        for (int ni = 0; ni < NT; ++ni) {
            int col = (gc * NT + ni) * 16 + l16;
            if (col < RO) {
                bf16_t* outp = xall + (size_t)(tm * 16) * RO + col;
                #pragma unroll
                for (int v = 0; v < 8; ++v) {
                    int m = v + half * 8;
                    outp[(size_t)m * RO] = (bf16_t)acc[mi][ni][v];
                }
            } else {
                int oc = col - RO;
                float bval = bias[oc];
                float* outp = rootout + (size_t)(tm * 16) * Odim + oc;
                #pragma unroll
                for (int v = 0; v < 8; ++v) {
                    int m = v + half * 8;
                    outp[(size_t)m * Odim] = acc[mi][ni][v] + bval;
                }
            }
        }
    }
}

// ---------------------------------------------------------------------------
// One wave per edge: agg[dst] += xall[src, etype] / cnt[dst, etype]
// (mean over (dst,rel) segment + sum over relations, fused via inv-count).
// PER = Odim/32 contiguous bf16 per lane, loaded as one vector.
// ---------------------------------------------------------------------------
template <int PER>
__global__ void __launch_bounds__(256)
scatter_edges_kernel(const int* __restrict__ ei, const int* __restrict__ et,
                     const float* __restrict__ inv,
                     const bf16_t* __restrict__ xall,
                     float* __restrict__ agg, int E) {
    const int Odim = PER * 32;
    int wid = blockIdx.x * (blockDim.x >> 5) + (threadIdx.x >> 5);
    if (wid >= E) return;
    int lane = threadIdx.x & 31;
    int src = ei[wid];
    int dst = ei[E + wid];
    int r   = et[wid];
    float w = inv[dst * R_REL + r];
    const bf16_t* m = xall + ((size_t)src * R_REL + r) * Odim + lane * PER;
    float* o = agg + (size_t)dst * Odim + lane * PER;
    bf16_t vals[PER];
    if constexpr (PER == 8) {
        v8bf t = *(const v8bf*)m;
        #pragma unroll
        for (int i = 0; i < 8; ++i) vals[i] = t[i];
    } else {
        v4bf t = *(const v4bf*)m;
        #pragma unroll
        for (int i = 0; i < 4; ++i) vals[i] = t[i];
    }
    #pragma unroll
    for (int i = 0; i < PER; ++i)
        __hip_atomic_fetch_add(o + i, (float)vals[i] * w,
                               __ATOMIC_RELAXED, __HIP_MEMORY_SCOPE_AGENT);
}

__global__ void leaky_bf16_kernel(const float* __restrict__ in,
                                  bf16_t* __restrict__ out, int n) {
    int i = blockIdx.x * blockDim.x + threadIdx.x;
    if (i >= n) return;
    float v = in[i];
    v = (v > 0.f) ? v : 0.01f * v;
    out[i] = (bf16_t)v;
}

// ---------------------------------------------------------------------------
extern "C" void kernel_launch(void* const* d_in, const int* in_sizes, int n_in,
                              void* d_out, int out_size, void* d_ws, size_t ws_size,
                              hipStream_t stream) {
    (void)in_sizes; (void)n_in; (void)out_size; (void)ws_size;
    const float* x      = (const float*)d_in[0];
    const int*   ei     = (const int*)  d_in[1];   // [2][E]
    const int*   et     = (const int*)  d_in[2];
    const float* bases1 = (const float*)d_in[3];
    const float* comp1  = (const float*)d_in[4];
    const float* root1  = (const float*)d_in[5];
    const float* bias1  = (const float*)d_in[6];
    const float* bases2 = (const float*)d_in[7];
    const float* comp2  = (const float*)d_in[8];
    const float* root2  = (const float*)d_in[9];
    const float* bias2  = (const float*)d_in[10];

    char* ws = (char*)d_ws;
    size_t off = 0;
    bf16_t* W1t    = (bf16_t*)(ws + off); off += (size_t)(R_REL + 1) * HID_DIM * IN_DIM * 2;  // 1.18 MB
    bf16_t* W2t    = (bf16_t*)(ws + off); off += (size_t)(R_REL + 1) * Z_DIM * HID_DIM * 2;   // 0.59 MB
    bf16_t* nodebf = (bf16_t*)(ws + off); off += (size_t)N_NODES * IN_DIM * 2;                // 15.4 MB (x, reused for z1)
    bf16_t* xall   = (bf16_t*)(ws + off); off += (size_t)N_NODES * R_REL * HID_DIM * 2;       // 123 MB (reused layer2)
    float*  cnt    = (float*) (ws + off); off += (size_t)N_NODES * R_REL * 4;                 // 0.96 MB
    float*  agg1   = (float*) (ws + off); off += (size_t)N_NODES * HID_DIM * 4;               // 30.7 MB
    float*  out    = (float*)d_out;

    // Weight prep (basis combine + transpose to [rel][o][k] bf16)
    prep_w_kernel<<<((R_REL + 1) * HID_DIM * IN_DIM + 255) / 256, 256, 0, stream>>>(
        bases1, comp1, root1, W1t, IN_DIM, HID_DIM);
    prep_w_kernel<<<((R_REL + 1) * Z_DIM * HID_DIM + 255) / 256, 256, 0, stream>>>(
        bases2, comp2, root2, W2t, HID_DIM, Z_DIM);
    cvt_bf16_kernel<<<(N_NODES * IN_DIM + 255) / 256, 256, 0, stream>>>(
        x, nodebf, N_NODES * IN_DIM);

    // Per-(dst, rel) edge counts -> inverse counts
    zero_f32_kernel<<<(N_NODES * R_REL + 255) / 256, 256, 0, stream>>>(cnt, N_NODES * R_REL);
    count_edges_kernel<<<(E_EDGES + 255) / 256, 256, 0, stream>>>(ei, et, cnt, E_EDGES);
    invert_cnt_kernel<<<(N_NODES * R_REL + 255) / 256, 256, 0, stream>>>(cnt, N_NODES * R_REL);

    const int groupsM = (N_NODES / 16 + MT - 1) / MT;   // 938

    // ---- Layer 1 ----
    {
        int tilesC  = ((R_REL + 1) * HID_DIM) / 16;     // 144
        int waves   = groupsM * (tilesC / NT);          // 938*36 = 33768
        gemm_wmma_kernel<<<(waves + 7) / 8, 256, 0, stream>>>(
            nodebf, W1t, xall, agg1, bias1, HID_DIM, tilesC);
        scatter_edges_kernel<8><<<(E_EDGES + 7) / 8, 256, 0, stream>>>(
            ei, et, cnt, xall, agg1, E_EDGES);
        leaky_bf16_kernel<<<(N_NODES * HID_DIM + 255) / 256, 256, 0, stream>>>(
            agg1, nodebf, N_NODES * HID_DIM);
    }

    // ---- Layer 2 (root+bias written straight into d_out, atomics on top) ----
    {
        int tilesC  = ((R_REL + 1) * Z_DIM) / 16;       // 72
        int waves   = groupsM * (tilesC / NT);          // 938*18 = 16884
        gemm_wmma_kernel<<<(waves + 7) / 8, 256, 0, stream>>>(
            nodebf, W2t, xall, out, bias2, Z_DIM, tilesC);
        scatter_edges_kernel<4><<<(E_EDGES + 7) / 8, 256, 0, stream>>>(
            ei, et, cnt, xall, out, E_EDGES);
    }
}
